// TrajGRUcell_53180285059612
// MI455X (gfx1250) — compile-verified, matrix operand order checked by reading, not run on hardware
//
#include <hip/hip_runtime.h>
#include <hip/hip_bf16.h>
#include <math.h>
#include <stdint.h>

// ---------------------------------------------------------------------------
// TrajGRU cell for MI455X (gfx1250, wave32, WMMA).
// All convs are lowered to bf16 im2col GEMMs executed with
// v_wmma_f32_16x16x32_bf16 (f32 accumulate). Weights are converted once per
// launch into [N][K] bf16 (natural OIHW flatten), activations are staged
// channels-last bf16 so GEMM A rows are contiguous. GEMM tiles are staged
// into LDS with GLOBAL_LOAD_ASYNC_TO_LDS_B128 when the toolchain exposes the
// builtin (ASYNCcnt-tracked, no VGPR round trip), else global->VGPR->ds.
// ---------------------------------------------------------------------------

typedef __bf16 bf16;
typedef bf16  v16bf __attribute__((ext_vector_type(16)));
typedef bf16  v8bf  __attribute__((ext_vector_type(8)));
typedef float v8f   __attribute__((ext_vector_type(8)));

#define LEAK 0.2f

// Problem dimensions (fixed by reference setup_inputs()).
#define BB   4
#define SS   6
#define CC   8
#define HH   96
#define WW   96
#define CH   64
#define LL   13
#define HWSZ (HH * WW)        // 9216
#define MQ   (BB * HWSZ)      // 36864  (divisible by 128)
#define M0Q  (BB * SS * HWSZ) // 221184 (divisible by 128)

#if defined(__has_builtin)
#if __has_builtin(__builtin_amdgcn_global_load_async_to_lds_b128)
#define HAS_ASYNC_LDS 1
#endif
#endif

#ifdef HAS_ASYNC_LDS
// Builtin signature (from hipcc diagnostic): param0 = v4i32 in AS1 (global),
// param1 = v4i32 in AS3 (LDS), then two constant ints (offset, cpol).
typedef int v4i __attribute__((ext_vector_type(4)));
typedef __attribute__((address_space(1))) v4i GV4;
typedef __attribute__((address_space(3))) v4i LV4;

__device__ __forceinline__ GV4* to_global(const void* p) {
  return (GV4*)(uintptr_t)p;
}
__device__ __forceinline__ LV4* to_lds(const void* p) {
  return (LV4*)(uint32_t)(uintptr_t)p;  // low 32 bits = LDS offset
}
__device__ __forceinline__ void wait_async0() {
#if __has_builtin(__builtin_amdgcn_s_wait_asynccnt)
  __builtin_amdgcn_s_wait_asynccnt(0);
#else
  asm volatile("s_wait_asynccnt 0x0" ::: "memory");
#endif
}
#endif

// ---------------------------------------------------------------------------
// WMMA fragment load: LDS tile stored [row][k] (32 bf16 per row). A lane needs
// K = {kh*8 .. kh*8+7} and {kh*8+16 .. kh*8+23} (16-bit A/B VGPR layout), i.e.
// two contiguous 16-byte chunks -> two ds_load_b128.
// ---------------------------------------------------------------------------
__device__ __forceinline__ v16bf load_frag(const bf16* p) {
  v8bf lo = *(const v8bf*)p;
  v8bf hi = *(const v8bf*)(p + 16);
  return __builtin_shufflevector(lo, hi, 0, 1, 2, 3, 4, 5, 6, 7,
                                         8, 9, 10, 11, 12, 13, 14, 15);
}

__device__ __forceinline__ void store_tile(float* __restrict__ C, int ldc,
                                           int Nout, int act,
                                           const float* __restrict__ bias,
                                           v8f acc, int mBase, int nBase,
                                           int lane) {
  // 16x16 f32 C layout: lane&15 = N, VGPR r -> M = (lane>>4)*8 + r
  int ln = lane & 15, mh = lane >> 4;
  int ng = nBase + ln;
  if (ng >= Nout) return;
  float bv = bias ? bias[ng] : 0.f;
#pragma unroll
  for (int r = 0; r < 8; ++r) {
    float v = acc[r] + bv;
    if (act) v = (v >= 0.f) ? v : LEAK * v;
    C[(size_t)(mBase + mh * 8 + r) * ldc + ng] = v;
  }
}

// ---------------------------------------------------------------------------
// Tiled bf16 GEMM: C[M x Nout] = A[M x K] * B^T (B stored [N][K]) + bias.
// Block tile 128(M) x 64(N); 8 waves of 32 lanes, wave tile 32x32 (2x2 WMMA).
// K must be a multiple of 32, gridDim.x*128 == M.
// ---------------------------------------------------------------------------
__global__ __launch_bounds__(256) void gemm_wmma_bf16(
    const bf16* __restrict__ A, const bf16* __restrict__ Bw,
    const float* __restrict__ bias, float* __restrict__ C, int K, int ldc,
    int Nout, int act) {
  __shared__ __attribute__((aligned(16))) bf16 ldsA[128 * 32];
  __shared__ __attribute__((aligned(16))) bf16 ldsB[64 * 32];

  const int tid  = threadIdx.x;
  const int m0   = blockIdx.x * 128;
  const int n0   = blockIdx.y * 64;
  const int lane = tid & 31;
  const int wave = tid >> 5;
  const int mW   = (wave >> 1) * 32;  // wave M offset in block tile
  const int nW   = (wave & 1) * 32;   // wave N offset in block tile
  const int lm   = lane & 15;
  const int kh   = lane >> 4;

  // A-tile cooperative load indices: 128 rows x 4 chunks(16B) = 512 chunks
  const int ra0 = tid >> 2, pa0 = tid & 3;
  const int ca1 = tid + 256;
  const int ra1 = ca1 >> 2, pa1 = ca1 & 3;
  // B-tile: 64 rows x 4 chunks = 256 chunks
  const int rb = tid >> 2, pb = tid & 3;

  v8f acc00 = {}, acc01 = {}, acc10 = {}, acc11 = {};

  for (int k0 = 0; k0 < K; k0 += 32) {
    __syncthreads();
#ifdef HAS_ASYNC_LDS
    // memory -> LDS without VGPR round trip; tracked on ASYNCcnt
    __builtin_amdgcn_global_load_async_to_lds_b128(
        to_global(A + (size_t)(m0 + ra0) * K + k0 + pa0 * 8),
        to_lds(&ldsA[ra0 * 32 + pa0 * 8]), 0, 0);
    __builtin_amdgcn_global_load_async_to_lds_b128(
        to_global(A + (size_t)(m0 + ra1) * K + k0 + pa1 * 8),
        to_lds(&ldsA[ra1 * 32 + pa1 * 8]), 0, 0);
    __builtin_amdgcn_global_load_async_to_lds_b128(
        to_global(Bw + (size_t)(n0 + rb) * K + k0 + pb * 8),
        to_lds(&ldsB[rb * 32 + pb * 8]), 0, 0);
#else
    *(uint4*)&ldsA[ra0 * 32 + pa0 * 8] =
        *(const uint4*)(A + (size_t)(m0 + ra0) * K + k0 + pa0 * 8);
    *(uint4*)&ldsA[ra1 * 32 + pa1 * 8] =
        *(const uint4*)(A + (size_t)(m0 + ra1) * K + k0 + pa1 * 8);
    *(uint4*)&ldsB[rb * 32 + pb * 8] =
        *(const uint4*)(Bw + (size_t)(n0 + rb) * K + k0 + pb * 8);
#endif
    if (k0 + 32 < K) {  // pull next K-slab of A toward the WGP (all levels)
      __builtin_prefetch(A + (size_t)(m0 + ra0) * K + k0 + 32 + pa0 * 8, 0, 3);
    }
#ifdef HAS_ASYNC_LDS
    wait_async0();
#endif
    __syncthreads();

    v16bf a0 = load_frag(&ldsA[(mW + lm) * 32 + kh * 8]);
    v16bf a1 = load_frag(&ldsA[(mW + 16 + lm) * 32 + kh * 8]);
    v16bf b0 = load_frag(&ldsB[(nW + lm) * 32 + kh * 8]);
    v16bf b1 = load_frag(&ldsB[(nW + 16 + lm) * 32 + kh * 8]);

    acc00 = __builtin_amdgcn_wmma_f32_16x16x32_bf16(false, a0, false, b0,
                                                    (short)0, acc00, false, false);
    acc01 = __builtin_amdgcn_wmma_f32_16x16x32_bf16(false, a0, false, b1,
                                                    (short)0, acc01, false, false);
    acc10 = __builtin_amdgcn_wmma_f32_16x16x32_bf16(false, a1, false, b0,
                                                    (short)0, acc10, false, false);
    acc11 = __builtin_amdgcn_wmma_f32_16x16x32_bf16(false, a1, false, b1,
                                                    (short)0, acc11, false, false);
  }

  store_tile(C, ldc, Nout, act, bias, acc00, m0 + mW,      n0 + nW,      lane);
  store_tile(C, ldc, Nout, act, bias, acc01, m0 + mW,      n0 + nW + 16, lane);
  store_tile(C, ldc, Nout, act, bias, acc10, m0 + mW + 16, n0 + nW,      lane);
  store_tile(C, ldc, Nout, act, bias, acc11, m0 + mW + 16, n0 + nW + 16, lane);
}

// ---------------------------------------------------------------------------
// Weight f32 (OIHW, rowsSrc x Ksrc flat) -> bf16 [rowsDst][Kpad], zero-padded.
// ---------------------------------------------------------------------------
__global__ void cvt_w(const float* __restrict__ src, bf16* __restrict__ dst,
                      int rowsDst, int Kpad, int rowsSrc, int Ksrc) {
  int idx = blockIdx.x * 256 + threadIdx.x;
  if (idx >= rowsDst * Kpad) return;
  int r = idx / Kpad, k = idx % Kpad;
  float v = (r < rowsSrc && k < Ksrc) ? src[(size_t)r * Ksrc + k] : 0.f;
  dst[idx] = (bf16)v;
}

__global__ void pad_b(const float* __restrict__ src, float* __restrict__ dst,
                      int n, int nsrc) {
  int i = blockIdx.x * 64 + threadIdx.x;
  if (i < n) dst[i] = (i < nsrc) ? src[i] : 0.f;
}

// ---------------------------------------------------------------------------
// im2col for the i2h 3x3 conv over all B*S frames (Cin=8 -> K=72 pad 96).
// ---------------------------------------------------------------------------
__global__ void im2col_i2h(const float* __restrict__ inp, bf16* __restrict__ A0) {
  size_t idx = (size_t)blockIdx.x * 256 + threadIdx.x;
  if (idx >= (size_t)M0Q * 96) return;
  int    k  = (int)(idx % 96);
  size_t m  = idx / 96;
  int    hw = (int)(m % HWSZ);
  int    bs = (int)(m / HWSZ);
  int    y = hw / WW, x = hw % WW;
  float  v = 0.f;
  if (k < 72) {
    int ci = k / 9, t = k % 9;
    int yy = y + t / 3 - 1, xx = x + t % 3 - 1;
    if (yy >= 0 && yy < HH && xx >= 0 && xx < WW)
      v = inp[((size_t)bs * CC + ci) * HWSZ + yy * WW + xx];
  }
  A0[idx] = (bf16)v;
}

// im2col for flow conv1: concat(x_t, h) 3x3, Cin=72 -> K=648 pad 672.
__global__ void im2col_flow1(const float* __restrict__ inp,
                             const float* __restrict__ h, int tstep,
                             bf16* __restrict__ A1) {
  size_t idx = (size_t)blockIdx.x * 256 + threadIdx.x;
  if (idx >= (size_t)MQ * 672) return;
  int    k  = (int)(idx % 672);
  size_t m  = idx / 672;
  int    hw = (int)(m % HWSZ);
  int    b  = (int)(m / HWSZ);
  int    y = hw / WW, x = hw % WW;
  float  v = 0.f;
  if (k < 648) {
    int ci = k / 9, t = k % 9;
    int yy = y + t / 3 - 1, xx = x + t % 3 - 1;
    if (yy >= 0 && yy < HH && xx >= 0 && xx < WW) {
      if (ci < CC)
        v = inp[(((size_t)b * SS + tstep) * CC + ci) * HWSZ + yy * WW + xx];
      else
        v = h[((size_t)b * CH + (ci - CC)) * HWSZ + yy * WW + xx];
    }
  }
  A1[idx] = (bf16)v;
}

// im2col for flow conv2: f (channels-last f32, ldc=32) 3x3, K=288.
__global__ void im2col_flow2(const float* __restrict__ f,
                             bf16* __restrict__ A2) {
  size_t idx = (size_t)blockIdx.x * 256 + threadIdx.x;
  if (idx >= (size_t)MQ * 288) return;
  int    k  = (int)(idx % 288);
  size_t m  = idx / 288;
  int    hw = (int)(m % HWSZ);
  int    b  = (int)(m / HWSZ);
  int    y = hw / WW, x = hw % WW;
  int    ci = k / 9, t = k % 9;
  int    yy = y + t / 3 - 1, xx = x + t % 3 - 1;
  float  v = 0.f;
  if (yy >= 0 && yy < HH && xx >= 0 && xx < WW)
    v = f[((size_t)b * HWSZ + yy * WW + xx) * 32 + ci];
  A2[idx] = (bf16)v;
}

// ---------------------------------------------------------------------------
// Bilinear warp of h by -flow (grid_sample, zero pad, align_corners=False);
// writes warped activations channels-last bf16: Aw[m*832 + l*64 + c].
// ix = (x - fx) * W/(W-1) - 0.5 (algebraic simplification of the reference).
// ---------------------------------------------------------------------------
__global__ void warp_bilinear(const float* __restrict__ h,
                              const float* __restrict__ flow,
                              bf16* __restrict__ Aw) {
  int idx = blockIdx.x * 256 + threadIdx.x;
  if (idx >= BB * LL * HWSZ) return;
  int x = idx % WW;
  int y = (idx / WW) % HH;
  int l = (idx / HWSZ) % LL;
  int b = idx / (LL * HWSZ);
  size_t m = (size_t)b * HWSZ + y * WW + x;
  float fx = flow[m * 32 + 2 * l];
  float fy = flow[m * 32 + 2 * l + 1];
  const float sx = (float)WW / (float)(WW - 1);
  const float sy = (float)HH / (float)(HH - 1);
  float ix = ((float)x - fx) * sx - 0.5f;
  float iy = ((float)y - fy) * sy - 0.5f;
  float x0f = floorf(ix), y0f = floorf(iy);
  int   x0 = (int)x0f, y0 = (int)y0f;
  float wx1 = ix - x0f, wy1 = iy - y0f;
  float wx0 = 1.f - wx1, wy0 = 1.f - wy1;
  int   x1 = x0 + 1, y1 = y0 + 1;
  bool vx0 = (x0 >= 0) && (x0 < WW), vx1 = (x1 >= 0) && (x1 < WW);
  bool vy0 = (y0 >= 0) && (y0 < HH), vy1 = (y1 >= 0) && (y1 < HH);
  int  xc0 = min(max(x0, 0), WW - 1), xc1 = min(max(x1, 0), WW - 1);
  int  yc0 = min(max(y0, 0), HH - 1), yc1 = min(max(y1, 0), HH - 1);
  float w00 = wx0 * wy0 * ((vx0 && vy0) ? 1.f : 0.f);
  float w10 = wx1 * wy0 * ((vx1 && vy0) ? 1.f : 0.f);
  float w01 = wx0 * wy1 * ((vx0 && vy1) ? 1.f : 0.f);
  float w11 = wx1 * wy1 * ((vx1 && vy1) ? 1.f : 0.f);
  int o00 = yc0 * WW + xc0, o10 = yc0 * WW + xc1;
  int o01 = yc1 * WW + xc0, o11 = yc1 * WW + xc1;
  const float* hb  = h + (size_t)b * CH * HWSZ;
  bf16*        dst = Aw + m * 832 + l * CH;
#pragma unroll 4
  for (int c = 0; c < CH; ++c) {
    const float* hc = hb + (size_t)c * HWSZ;
    float v = hc[o00] * w00 + hc[o10] * w10 + hc[o01] * w01 + hc[o11] * w11;
    dst[c] = (bf16)v;
  }
}

// ---------------------------------------------------------------------------
// GRU gate fusion: r/u sigmoid, leaky candidate, state blend; writes the new
// state (channels-first) and the timestep slice of the output tensor.
// ---------------------------------------------------------------------------
__global__ void gru_gate(const float* __restrict__ i2h,
                         const float* __restrict__ h2h,
                         const float* __restrict__ hold,
                         float* __restrict__ hnew, float* __restrict__ out,
                         int tstep) {
  int idx = blockIdx.x * 256 + threadIdx.x;
  if (idx >= MQ * CH) return;
  int c  = idx & 63;
  int m  = idx >> 6;
  int hw = m % HWSZ;
  int b  = m / HWSZ;
  size_t mi = ((size_t)b * SS + tstep) * HWSZ + hw;  // i2h row (channels-last)
  float i1  = i2h[mi * 192 + c];
  float i2v = i2h[mi * 192 + 64 + c];
  float i3  = i2h[mi * 192 + 128 + c];
  float h1  = h2h[(size_t)m * 192 + c];
  float h2v = h2h[(size_t)m * 192 + 64 + c];
  float h3  = h2h[(size_t)m * 192 + 128 + c];
  float hp  = hold[((size_t)b * CH + c) * HWSZ + hw];
  float r   = 1.f / (1.f + __expf(-(i1 + h1)));
  float u   = 1.f / (1.f + __expf(-(i2v + h2v)));
  float mm  = i3 + r * h3;
  mm = (mm >= 0.f) ? mm : LEAK * mm;
  float nh = u * hp + (1.f - u) * mm;
  hnew[((size_t)b * CH + c) * HWSZ + hw] = nh;
  out[(((size_t)b * SS + tstep) * CH + c) * HWSZ + hw] = nh;
}

// ---------------------------------------------------------------------------
extern "C" void kernel_launch(void* const* d_in, const int* in_sizes, int n_in,
                              void* d_out, int out_size, void* d_ws,
                              size_t ws_size, hipStream_t stream) {
  const float* inp    = (const float*)d_in[0];  // (B,S,8,96,96)
  const float* states = (const float*)d_in[1];  // (B,64,96,96)
  const float* fw1    = (const float*)d_in[2];  // (32,72,3,3)
  const float* fb1    = (const float*)d_in[3];
  const float* fw2    = (const float*)d_in[4];  // (26,32,3,3)
  const float* fb2    = (const float*)d_in[5];
  const float* i2h_w  = (const float*)d_in[6];  // (192,8,3,3)
  const float* i2h_b  = (const float*)d_in[7];
  const float* ret_w  = (const float*)d_in[8];  // (192,832,1,1)
  const float* ret_b  = (const float*)d_in[9];
  float*       out    = (float*)d_out;          // (B,S,64,96,96) ++ (B,64,96,96)

  // -------- workspace carve-out (256B aligned) --------
  char*  ws  = (char*)d_ws;
  size_t off = 0;
  auto carve = [&](size_t bytes) -> char* {
    char* p = ws + off;
    off = (off + bytes + 255) & ~(size_t)255;
    return p;
  };
  bf16*  w0t  = (bf16*)carve((size_t)192 * 96 * 2);   // i2h weights  [192][96]
  bf16*  w1t  = (bf16*)carve((size_t)64 * 672 * 2);   // flow1 weights[64][672]
  bf16*  w2t  = (bf16*)carve((size_t)64 * 288 * 2);   // flow2 weights[64][288]
  bf16*  w3t  = (bf16*)carve((size_t)192 * 832 * 2);  // ret weights  [192][832]
  float* b1p  = (float*)carve(64 * 4);
  float* b2p  = (float*)carve(64 * 4);
  bf16*  A0   = (bf16*)carve((size_t)M0Q * 96 * 2);   // i2h im2col
  float* i2h  = (float*)carve((size_t)M0Q * 192 * 4); // i2h result (ch-last)
  bf16*  A1   = (bf16*)carve((size_t)MQ * 672 * 2);   // flow1 im2col
  float* fbuf = (float*)carve((size_t)MQ * 32 * 4);   // f (ch-last, leaky)
  bf16*  A2   = (bf16*)carve((size_t)MQ * 288 * 2);   // flow2 im2col
  float* flwb = (float*)carve((size_t)MQ * 32 * 4);   // flows (ch-last, ldc 32)
  bf16*  Aw   = (bf16*)carve((size_t)MQ * 832 * 2);   // warped (ch-last bf16)
  float* h2h  = (float*)carve((size_t)MQ * 192 * 4);  // ret conv out
  float* hA   = (float*)carve((size_t)BB * CH * HWSZ * 4);
  float* hB   = (float*)carve((size_t)BB * CH * HWSZ * 4);

  // -------- one-time weight/bias conversion --------
  cvt_w<<<(192 * 96 + 255) / 256, 256, 0, stream>>>(i2h_w, w0t, 192, 96, 192, 72);
  cvt_w<<<(64 * 672 + 255) / 256, 256, 0, stream>>>(fw1, w1t, 64, 672, 32, 648);
  cvt_w<<<(64 * 288 + 255) / 256, 256, 0, stream>>>(fw2, w2t, 64, 288, 26, 288);
  cvt_w<<<(192 * 832 + 255) / 256, 256, 0, stream>>>(ret_w, w3t, 192, 832, 192, 832);
  pad_b<<<1, 64, 0, stream>>>(fb1, b1p, 64, 32);
  pad_b<<<1, 64, 0, stream>>>(fb2, b2p, 64, 26);

  // -------- i2h precompute over all timesteps --------
  im2col_i2h<<<(unsigned)((size_t)M0Q * 96 / 256), 256, 0, stream>>>(inp, A0);
  gemm_wmma_bf16<<<dim3(M0Q / 128, 3), 256, 0, stream>>>(A0, w0t, i2h_b, i2h,
                                                         96, 192, 192, 0);

  // -------- recurrent loop --------
  for (int t = 0; t < SS; ++t) {
    const float* hcur = (t == 0) ? states : ((t & 1) ? hA : hB);
    float*       hnxt = (t & 1) ? hB : hA;

    im2col_flow1<<<(unsigned)((size_t)MQ * 672 / 256), 256, 0, stream>>>(
        inp, hcur, t, A1);
    gemm_wmma_bf16<<<dim3(MQ / 128, 1), 256, 0, stream>>>(A1, w1t, b1p, fbuf,
                                                          672, 32, 32, 1);
    im2col_flow2<<<(unsigned)((size_t)MQ * 288 / 256), 256, 0, stream>>>(fbuf,
                                                                         A2);
    gemm_wmma_bf16<<<dim3(MQ / 128, 1), 256, 0, stream>>>(A2, w2t, b2p, flwb,
                                                          288, 32, 26, 0);
    warp_bilinear<<<(BB * LL * HWSZ + 255) / 256, 256, 0, stream>>>(hcur, flwb,
                                                                    Aw);
    gemm_wmma_bf16<<<dim3(MQ / 128, 3), 256, 0, stream>>>(Aw, w3t, ret_b, h2h,
                                                          832, 192, 192, 0);
    gru_gate<<<(MQ * CH + 255) / 256, 256, 0, stream>>>(i2h, h2h, hcur, hnxt,
                                                        out, t);
  }

  // last hidden state appended after the (B,S,Ch,H,W) outputs
  const float* hlast = ((SS - 1) & 1) ? hB : hA;
  (void)hipMemcpyAsync(out + (size_t)BB * SS * CH * HWSZ, hlast,
                       sizeof(float) * BB * CH * HWSZ,
                       hipMemcpyDeviceToDevice, stream);
}
